// RelativeMultiHeadAttention_50148038148292
// MI455X (gfx1250) — matrix operand order repaired and checked by reading, not compiled, not run on hardware
//
#include <hip/hip_runtime.h>
#include <hip/hip_bf16.h>
#include <hip/hip_fp16.h>

// ---------------------------------------------------------------------------
// RelativeMultiHeadAttention on MI455X (gfx1250, wave32, WMMA).
//
//   (0) fp32 -> f16 conversion of activations; weights converted AND
//       transposed to [N][K] so LDS staging is contiguous 16B copies.
//   (1) q/k/v/p projections: f16 WMMA GEMM, M=16384, N=K=1024.
//       Each wave computes a 16x64 strip (4 accumulators) so one A fragment
//       feeds 4 WMMAs. Weights are staged per 256-deep K chunk into LDS with
//       DOUBLE-BUFFERED GLOBAL_LOAD_ASYNC_TO_LDS_B128: chunk i+1's async DMA
//       overlaps chunk i's WMMAs; s_wait_asynccnt(8) drains only the current
//       chunk (FIFO per-wave ASYNCcnt).
//   (2) per-(b,t) HxH attention: WMMA scores + LDS softmax + WMMA context,
//       written in the reference's transpose/reshape layout.
//   (3) output GEMM ctx @ Wo + bo -> fp32 d_out.
// ---------------------------------------------------------------------------

typedef __attribute__((ext_vector_type(16))) _Float16 v16h;
typedef __attribute__((ext_vector_type(8)))  _Float16 v8h;
typedef __attribute__((ext_vector_type(8)))  float    v8f;
typedef __attribute__((ext_vector_type(4)))  int      v4i;

#define B_  8
#define T_  2048
#define DM  1024
#define H_  16
#define DH  64
#define M_  (B_ * T_)   // 16384
#define KC  256         // K-chunk staged in LDS per iteration
#define NCHUNK (DM / KC)

// --- async global->LDS staging path (guarded; flip to 0 if toolchain differs)
#define ASYNC_TRY 1
#if ASYNC_TRY && defined(__has_builtin)
#if __has_builtin(__builtin_amdgcn_global_load_async_to_lds_b128) && \
    __has_builtin(__builtin_amdgcn_s_wait_asynccnt)
#define ASYNC_STAGE 1
#endif
#endif
#ifndef ASYNC_STAGE
#define ASYNC_STAGE 0
#endif

#define AS1 __attribute__((address_space(1)))
#define AS3 __attribute__((address_space(3)))

// ---------------------------- fp32 -> f16 ----------------------------------
__global__ void f32_to_f16_kernel(const float* __restrict__ src,
                                  _Float16* __restrict__ dst, int n) {
  int i = blockIdx.x * blockDim.x + threadIdx.x;
  int stride = gridDim.x * blockDim.x;
  for (; i < n; i += stride) dst[i] = (_Float16)src[i];
}

// ------------------- fp32 [K][N] -> f16 transposed [N][K] ------------------
__global__ void transpose_f32_to_f16_kernel(const float* __restrict__ src,
                                            _Float16* __restrict__ dst) {
  int i = blockIdx.x * blockDim.x + threadIdx.x;
  int stride = gridDim.x * blockDim.x;
  for (; i < DM * DM; i += stride) {
    int n = i / DM;          // contiguous writes along k
    int k = i % DM;
    dst[i] = (_Float16)src[(size_t)k * DM + n];
  }
}

// ---------------------------- WMMA GEMM ------------------------------------
// C[M,N] = A[M,K] * W[K,N] (+bias), W given transposed as Wt[N][K] f16.
// 256 threads = 8 waves. Block computes 128 rows x 64 cols.
// Double-buffered LDS (2 x 32 KB): async-stage chunk ci+1 while computing ci.
// grid = (N/64, M/128) = (16, 128).
__global__ __launch_bounds__(256)
void gemm_wmma(const _Float16* __restrict__ A, const _Float16* __restrict__ Wt,
               const float* __restrict__ bias,
               _Float16* __restrict__ outH, float* __restrict__ outF) {
  __shared__ _Float16 ldsB[2][64 * KC];   // [buf][n][k]

  const int tid  = threadIdx.x;
  const int lane = tid & 31;
  const int wave = tid >> 5;
  const int n0   = blockIdx.x * 64;
  const int row0 = (blockIdx.y * 8 + wave) * 16;
  const int half = lane >> 4;
  const int nl   = lane & 15;

  const _Float16* ap = A + (size_t)(row0 + nl) * DM;   // A row for this lane

  // Stage Wt[n0..n0+64)[kc..kc+KC) into ldsB[buf]: 8 x B128 per thread.
  auto stage = [&](int buf, int kc) {
    const _Float16* gsrc = Wt + (size_t)n0 * DM + kc;
    _Float16* dbase = &ldsB[buf][0];
#pragma unroll
    for (int g = 0; g < 8; ++g) {                 // (64*KC/8)/256 = 8 granules
      int i = tid + g * 256;
      int n = i >> 5;                             // KC/8 = 32 granules per row
      int c = i & 31;
      const _Float16* gp = gsrc + (size_t)n * DM + c * 8;
      _Float16*       lp = dbase + n * KC + c * 8;
#if ASYNC_STAGE
      __builtin_amdgcn_global_load_async_to_lds_b128(
          (AS1 v4i*)gp, (AS3 v4i*)lp, 0, 0);
#else
      *(v8h*)lp = *(const v8h*)gp;
#endif
    }
  };

  v8f acc[4] = {v8f{}, v8f{}, v8f{}, v8f{}};

  stage(0, 0);                                    // prologue: chunk 0 in flight

#pragma unroll
  for (int ci = 0; ci < NCHUNK; ++ci) {
    const int cur = ci & 1;
    const int kc  = ci * KC;

    if (ci + 1 < NCHUNK) {
      stage(1 - cur, kc + KC);                    // overlap next chunk's DMA
#if ASYNC_STAGE
      __builtin_amdgcn_s_wait_asynccnt(8);        // drain current chunk only
#endif
    } else {
#if ASYNC_STAGE
      __builtin_amdgcn_s_wait_asynccnt(0);
#endif
    }
    __syncthreads();                              // chunk ci visible block-wide

    const _Float16* lb = &ldsB[cur][0];
#pragma unroll
    for (int k0 = 0; k0 < KC; k0 += 32) {
      __builtin_prefetch(ap + kc + k0 + 512, 0, 1);      // global_prefetch_b8

      // A fragment (16-bit A 16x32): elems 0-7 -> K=half*8+j, 8-15 -> +16.
      v8h a0 = *(const v8h*)(ap + kc + k0 + half * 8);
      v8h a1 = *(const v8h*)(ap + kc + k0 + 16 + half * 8);
      v16h a;
#pragma unroll
      for (int j = 0; j < 8; ++j) { a[j] = a0[j]; a[8 + j] = a1[j]; }

#pragma unroll
      for (int nt = 0; nt < 4; ++nt) {
        // B fragment (32x16): elems 0-7 -> K=k0+half*8+j, 8-15 -> +16, N=nt*16+nl.
        const _Float16* bp = lb + (nt * 16 + nl) * KC + k0 + half * 8;
        v8h b0 = *(const v8h*)bp;
        v8h b1 = *(const v8h*)(bp + 16);
        v16h b;
#pragma unroll
        for (int j = 0; j < 8; ++j) { b[j] = b0[j]; b[8 + j] = b1[j]; }

        acc[nt] = __builtin_amdgcn_wmma_f32_16x16x32_f16(
            false, a, false, b, (short)0, acc[nt], false, false);
      }
    }
    __syncthreads();                              // done reading ldsB[cur]
  }

  // ---- epilogue: bias + store 4 tiles
#pragma unroll
  for (int nt = 0; nt < 4; ++nt) {
    const int col = n0 + nt * 16 + nl;
    const float bval = bias ? bias[col] : 0.0f;
#pragma unroll
    for (int r = 0; r < 8; ++r) {
      size_t row = (size_t)row0 + half * 8 + r;   // C layout: VGPR r, col nl
      float v = acc[nt][r] + bval;
      if (outH) outH[row * DM + col] = (_Float16)v;
      else      outF[row * DM + col] = v;
    }
  }
}

// ---------------------------- attention ------------------------------------
// One wave per (b,t). rel_shift: out position j reads its pos-score from flat
// index f = j + B over a (B, T+1) zero-front-padded layout:
//   b_s = f/(T+1), t_s = f%(T+1); pos contribution is zero when t_s == 0.
__global__ __launch_bounds__(256)
void attn_wmma(const _Float16* __restrict__ q16, const _Float16* __restrict__ k16,
               const _Float16* __restrict__ vv16, const _Float16* __restrict__ p16,
               const float* __restrict__ u_bias, const float* __restrict__ v_bias,
               const unsigned char* __restrict__ masks,
               _Float16* __restrict__ ctx16) {
  __shared__ float    sc[8][16 * 16];   // score tiles, one per wave
  __shared__ _Float16 at[8][16 * 32];   // attn, zero-padded to K=32

  const int tid  = threadIdx.x;
  const int lane = tid & 31;
  const int wave = tid >> 5;
  const int half = lane >> 4;
  const int nl   = lane & 15;

  const int j = blockIdx.x * 8 + wave;     // flat (b,t) position
  const int b = j / T_;
  const int t = j % T_;

  const int f  = j + B_;
  const int bs = f / (T_ + 1);
  const int ts = f % (T_ + 1);
  const bool pvalid = (ts > 0) && (bs < B_);
  const int  src    = bs * T_ + (ts - 1);

  // ---- scores: content (q+u)k^T plus shifted positional (q'+v)p'^T --------
  v8f acc = {};
  {
    const _Float16* qp = q16 + (size_t)j * DM;
    const _Float16* kp = k16 + (size_t)j * DM;
#pragma unroll
    for (int kc = 0; kc < DH; kc += 32) {
      v16h a, bb;
      const _Float16* aq = qp + nl * DH + kc;          // A row = head nl
      const float*    uu = u_bias + nl * DH + kc;
#pragma unroll
      for (int e = 0; e < 8; ++e) {
        a[e]     = (_Float16)((float)aq[half * 8 + e]      + uu[half * 8 + e]);
        a[8 + e] = (_Float16)((float)aq[16 + half * 8 + e] + uu[16 + half * 8 + e]);
      }
      const _Float16* bk = kp + nl * DH + kc + half * 8;  // B col = head nl
#pragma unroll
      for (int e = 0; e < 16; ++e) bb[e] = bk[(e & 7) + (e >> 3) * 16];
      acc = __builtin_amdgcn_wmma_f32_16x16x32_f16(false, a, false, bb,
                                                   (short)0, acc, false, false);
    }
  }
  if (pvalid) {   // wave-uniform branch: EXEC stays all-ones for WMMA
    const _Float16* qp = q16 + (size_t)src * DM;
    const _Float16* pp = p16 + (size_t)src * DM;
#pragma unroll
    for (int kc = 0; kc < DH; kc += 32) {
      v16h a, bb;
      const _Float16* aq = qp + nl * DH + kc;
      const float*    vv = v_bias + nl * DH + kc;
#pragma unroll
      for (int e = 0; e < 8; ++e) {
        a[e]     = (_Float16)((float)aq[half * 8 + e]      + vv[half * 8 + e]);
        a[8 + e] = (_Float16)((float)aq[16 + half * 8 + e] + vv[16 + half * 8 + e]);
      }
      const _Float16* bp = pp + nl * DH + kc + half * 8;
#pragma unroll
      for (int e = 0; e < 16; ++e) bb[e] = bp[(e & 7) + (e >> 3) * 16];
      acc = __builtin_amdgcn_wmma_f32_16x16x32_f16(false, a, false, bb,
                                                   (short)0, acc, false, false);
    }
  }

  // ---- scale + mask, spill C tile to LDS for the row softmax --------------
  const bool mk = masks[j] != 0;
  const float scale = 0.03125f;   // 1/sqrt(1024)
#pragma unroll
  for (int r = 0; r < 8; ++r) {
    float s = mk ? acc[r] * scale : 1e-9f;
    sc[wave][(half * 8 + r) * 16 + nl] = s;
  }
  __syncthreads();

  // ---- softmax over the 16 key-heads, emit f16 attn zero-padded to K=32 ---
  if (lane < 16) {
    float* row = &sc[wave][lane * 16];
    float mx = row[0];
#pragma unroll
    for (int c = 1; c < 16; ++c) mx = fmaxf(mx, row[c]);
    float e[16], sum = 0.0f;
#pragma unroll
    for (int c = 0; c < 16; ++c) { e[c] = __expf(row[c] - mx); sum += e[c]; }
    const float inv = 1.0f / sum;
    _Float16* arow = &at[wave][lane * 32];
#pragma unroll
    for (int c = 0; c < 16; ++c) arow[c] = (_Float16)(e[c] * inv);
#pragma unroll
    for (int c = 16; c < 32; ++c) arow[c] = (_Float16)0.0f;
  }
  __syncthreads();

  // ---- ctx = attn(16x16, padded to K=32) @ v(16x64) -----------------------
  v16h aat;
  {
    const _Float16* ar = &at[wave][nl * 32];
#pragma unroll
    for (int e = 0; e < 8; ++e) {
      aat[e]     = ar[half * 8 + e];
      aat[8 + e] = ar[16 + half * 8 + e];   // zero-padded region
    }
  }
  const _Float16* vp = vv16 + (size_t)j * DM;
#pragma unroll
  for (int ntile = 0; ntile < 4; ++ntile) {
    const int n0 = ntile * 16;
    v16h bb;
#pragma unroll
    for (int e = 0; e < 16; ++e) {
      int kk = half * 8 + (e & 7) + (e >> 3) * 16;  // rows 16..31 are the zero pad
      bb[e] = (kk < 16) ? vp[kk * DH + n0 + nl] : (_Float16)0.0f;
    }
    v8f c2 = {};
    c2 = __builtin_amdgcn_wmma_f32_16x16x32_f16(false, aat, false, bb,
                                                (short)0, c2, false, false);
    // Store directly in the reference's transpose(0,2,1,3)+reshape layout:
    // row = h*128 + t/16, col = (t%16)*64 + d  within (B, 2048, 1024).
#pragma unroll
    for (int r = 0; r < 8; ++r) {
      int h = half * 8 + r;
      int d = n0 + nl;
      size_t row = (size_t)h * 128 + (t >> 4);
      size_t col = (size_t)(t & 15) * 64 + d;
      ctx16[((size_t)b * 2048 + row) * DM + col] = (_Float16)c2[r];
    }
  }
}

// ---------------------------- launcher -------------------------------------
extern "C" void kernel_launch(void* const* d_in, const int* in_sizes, int n_in,
                              void* d_out, int out_size, void* d_ws, size_t ws_size,
                              hipStream_t stream) {
  const float* inputs = (const float*)d_in[0];
  const float* pos    = (const float*)d_in[1];
  const unsigned char* masks = (const unsigned char*)d_in[2];
  const float* Wq = (const float*)d_in[3];
  const float* bq = (const float*)d_in[4];
  const float* Wk = (const float*)d_in[5];
  const float* bk = (const float*)d_in[6];
  const float* Wv = (const float*)d_in[7];
  const float* bv = (const float*)d_in[8];
  const float* Wp = (const float*)d_in[9];
  const float* Wo = (const float*)d_in[10];
  const float* bo = (const float*)d_in[11];
  const float* ub = (const float*)d_in[12];
  const float* vb = (const float*)d_in[13];

  // Workspace carve-up (f16 arena, 256B-aligned slices).
  char* ws = (char*)d_ws;
  size_t off = 0;
  auto carve = [&](size_t bytes) -> void* {
    void* p = ws + off;
    off += (bytes + 255) & ~(size_t)255;
    return p;
  };
  const size_t actB = (size_t)M_ * DM * sizeof(_Float16);   // 32 MB
  const size_t wB   = (size_t)DM * DM * sizeof(_Float16);   // 2 MB
  _Float16* in16  = (_Float16*)carve(actB);
  _Float16* pos16 = (_Float16*)carve(actB);
  _Float16* wq16  = (_Float16*)carve(wB);     // transposed [N][K]
  _Float16* wk16  = (_Float16*)carve(wB);
  _Float16* wv16  = (_Float16*)carve(wB);
  _Float16* wp16  = (_Float16*)carve(wB);
  _Float16* wo16  = (_Float16*)carve(wB);
  _Float16* q16   = (_Float16*)carve(actB);
  _Float16* k16   = (_Float16*)carve(actB);
  _Float16* v16w  = (_Float16*)carve(actB);
  _Float16* p16   = (_Float16*)carve(actB);
  _Float16* ctx16 = (_Float16*)carve(actB);

  const int nAct = M_ * DM;     // 16.8M
  f32_to_f16_kernel<<<1024, 256, 0, stream>>>(inputs, in16, nAct);
  f32_to_f16_kernel<<<1024, 256, 0, stream>>>(pos,    pos16, nAct);
  transpose_f32_to_f16_kernel<<<256, 256, 0, stream>>>(Wq, wq16);
  transpose_f32_to_f16_kernel<<<256, 256, 0, stream>>>(Wk, wk16);
  transpose_f32_to_f16_kernel<<<256, 256, 0, stream>>>(Wv, wv16);
  transpose_f32_to_f16_kernel<<<256, 256, 0, stream>>>(Wp, wp16);
  transpose_f32_to_f16_kernel<<<256, 256, 0, stream>>>(Wo, wo16);

  dim3 gemmGrid(DM / 64, M_ / 128);   // (16, 128)
  gemm_wmma<<<gemmGrid, 256, 0, stream>>>(in16,  wq16, bq, q16,  nullptr);
  gemm_wmma<<<gemmGrid, 256, 0, stream>>>(in16,  wk16, bk, k16,  nullptr);
  gemm_wmma<<<gemmGrid, 256, 0, stream>>>(in16,  wv16, bv, v16w, nullptr);
  gemm_wmma<<<gemmGrid, 256, 0, stream>>>(pos16, wp16, nullptr, p16, nullptr);

  attn_wmma<<<M_ / 8, 256, 0, stream>>>(q16, k16, v16w, p16, ub, vb, masks, ctx16);

  gemm_wmma<<<gemmGrid, 256, 0, stream>>>(ctx16, wo16, bo, nullptr, (float*)d_out);
}
